// Attention_63891933495367
// MI455X (gfx1250) — compile-verified, matrix operand order checked by reading, not run on hardware
//
#include <hip/hip_runtime.h>
#include <math.h>

typedef __bf16 bf16;
typedef __attribute__((ext_vector_type(8)))  __bf16 v8bf;
typedef __attribute__((ext_vector_type(16))) __bf16 v16bf;
typedef __attribute__((ext_vector_type(8)))  float  v8f;
typedef unsigned int u32x4 __attribute__((ext_vector_type(4)));
typedef int          i32x4 __attribute__((ext_vector_type(4)));
typedef int          i32x8 __attribute__((ext_vector_type(8)));

#define NUM_HEADS 16
#define NUM_KV_HEADS 4
#define HEAD_DIM 128
#define SEQ 2048
#define BATCH 2
#define HID 2048
#define MROWS (BATCH * SEQ)   // 4096

// ---------- helpers ----------
__device__ __forceinline__ bf16 f2bf(float x) {
  union { float f; unsigned u; } a; a.f = x;
  unsigned r = a.u + 0x7FFFu + ((a.u >> 16) & 1u);   // round-to-nearest-even
  union { unsigned short s; bf16 b; } c;
  c.s = (unsigned short)(r >> 16);
  return c.b;
}

// Build a 16x32 (or 32x16) bf16 WMMA fragment from two 16-byte runs.
__device__ __forceinline__ v16bf load_frag(const bf16* p0, const bf16* p1) {
  v8bf lo = *(const v8bf*)p0;
  v8bf hi = *(const v8bf*)p1;
  v16bf r;
#pragma unroll
  for (int i = 0; i < 8; ++i) { r[i] = lo[i]; r[i + 8] = hi[i]; }
  return r;
}

// TDM: DMA a 32(rows) x 128(bf16) tile, row stride 128 elems, global -> LDS.
// D# per CDNA5 ISA ch.8: group0 = {flags, lds_addr, global_addr, type=2},
// group1 = {data_size=2B, tensor dims 128x32, tile dims 128x32, stride0=128}.
// This toolchain exposes the 6-arg builtin (g0, g1, g2, g3, g4, cpol).
__device__ __forceinline__ void tdm_load_tile_32x128(const bf16* gptr, unsigned lds_off) {
  unsigned long long ga = (unsigned long long)(size_t)gptr;
  u32x4 g0;
  g0[0] = 1u;                                        // count=1, load, user mode
  g0[1] = lds_off;                                   // lds_addr
  g0[2] = (unsigned)(ga & 0xFFFFFFFFu);              // global_addr[31:0]
  g0[3] = (unsigned)((ga >> 32) & 0x1FFFFFFu)        // global_addr[56:32]
        | (2u << 30);                                // type = 2 ("image")
  i32x8 g1;
  g1[0] = (int)(1u << 16);                           // data_size=1 (2 bytes)
  g1[1] = (int)(128u << 16);                         // tensor_dim0 lo16 (=128)
  g1[2] = (int)(32u << 16);                          // dim0 hi16=0 | tensor_dim1 lo16 (=32)
  g1[3] = (int)(128u << 16);                         // dim1 hi16=0 | tile_dim0 (=128)
  g1[4] = 32;                                        // tile_dim1=32 | tile_dim2=0
  g1[5] = 128;                                       // tensor_dim0_stride[31:0] = 128
  g1[6] = 0;                                         // stride0 hi16 | stride1 lo16
  g1[7] = 0;                                         // stride1 hi32
  i32x4 g2 = {0, 0, 0, 0};
  i32x4 g3 = {0, 0, 0, 0};
  i32x8 g4 = {0, 0, 0, 0, 0, 0, 0, 0};
  __builtin_amdgcn_tensor_load_to_lds(g0, g1, g2, g3, g4, 0);
}

// ---------- f32 -> bf16 convert ----------
__global__ void cvt_f32_bf16(const float* __restrict__ src, bf16* __restrict__ dst, int n) {
  int i = blockIdx.x * blockDim.x + threadIdx.x;
  if (i < n) dst[i] = f2bf(src[i]);
}

// ---------- GEMM: C[M,N] = A[M,K](bf16,row) * Bw[N,K](bf16,row)^T, f32 out ----------
// One wave computes a 64x64 tile via 4x4 WMMA frags: 16 WMMA per 16 b128 loads.
__global__ __launch_bounds__(256) void gemm_bf16(const bf16* __restrict__ A,
                                                 const bf16* __restrict__ Bw,
                                                 float* __restrict__ C,
                                                 int M, int N, int K) {
  int wave = (blockIdx.x * blockDim.x + threadIdx.x) >> 5;
  int lane = threadIdx.x & 31;
  int tilesN = N >> 6;
  int m0 = (wave / tilesN) << 6;
  int n0 = (wave % tilesN) << 6;
  int l  = lane & 15;
  int hf = lane >> 4;

  v8f acc[4][4] = {};
  const bf16* pa = A  + (size_t)(m0 + l) * K + hf * 8;
  const bf16* pb = Bw + (size_t)(n0 + l) * K + hf * 8;

#pragma unroll 1
  for (int k0 = 0; k0 < K; k0 += 32) {
    v16bf a[4], b[4];
#pragma unroll
    for (int i = 0; i < 4; ++i) {
      const bf16* p = pa + (size_t)(16 * i) * K + k0;
      const bf16* q = pb + (size_t)(16 * i) * K + k0;
      a[i] = load_frag(p, p + 16);
      b[i] = load_frag(q, q + 16);
    }
#pragma unroll
    for (int i = 0; i < 4; ++i)
#pragma unroll
      for (int j = 0; j < 4; ++j)
        acc[i][j] = __builtin_amdgcn_wmma_f32_16x16x32_bf16(false, a[i], false, b[j],
                                                            (short)0, acc[i][j], false, false);
  }
#pragma unroll
  for (int i = 0; i < 4; ++i)
#pragma unroll
    for (int j = 0; j < 4; ++j)
#pragma unroll
      for (int r = 0; r < 8; ++r)
        C[(size_t)(m0 + 16 * i + r + hf * 8) * N + n0 + 16 * j + l] = acc[i][j][r];
}

// ---------- RMSNorm + RoPE (+ scale fold), f32 [tok, h*128+d] -> bf16 [b,h,s,d] ----------
__global__ __launch_bounds__(256) void norm_rope(const float* __restrict__ src,
                                                 const float* __restrict__ w,
                                                 const float* __restrict__ cosp,
                                                 const float* __restrict__ sinp,
                                                 bf16* __restrict__ dst,
                                                 int nheads, float outscale) {
  int wave = (blockIdx.x * blockDim.x + threadIdx.x) >> 5;
  int lane = threadIdx.x & 31;
  int head = wave % nheads;
  int tok  = wave / nheads;          // b*SEQ + s
  int b = tok / SEQ, s = tok % SEQ;

  const float* x = src + (size_t)tok * (nheads * HEAD_DIM) + head * HEAD_DIM;
  float v0 = x[lane], v1 = x[lane + 32], v2 = x[lane + 64], v3 = x[lane + 96];

  float ss = v0 * v0 + v1 * v1 + v2 * v2 + v3 * v3;
#pragma unroll
  for (int m = 16; m >= 1; m >>= 1) ss += __shfl_xor(ss, m, 32);
  float rinv = rsqrtf(ss * (1.0f / 128.0f) + 1e-6f);
  v0 *= rinv * w[lane];      v1 *= rinv * w[lane + 32];
  v2 *= rinv * w[lane + 64]; v3 *= rinv * w[lane + 96];

  const float* cb = cosp + (size_t)tok * HEAD_DIM;
  const float* sb = sinp + (size_t)tok * HEAD_DIM;
  float o0 = v0 * cb[lane]      - v2 * sb[lane];
  float o1 = v1 * cb[lane + 32] - v3 * sb[lane + 32];
  float o2 = v2 * cb[lane + 64] + v0 * sb[lane + 64];
  float o3 = v3 * cb[lane + 96] + v1 * sb[lane + 96];

  bf16* d = dst + ((size_t)(b * nheads + head) * SEQ + s) * HEAD_DIM;
  d[lane]      = f2bf(o0 * outscale);
  d[lane + 32] = f2bf(o1 * outscale);
  d[lane + 64] = f2bf(o2 * outscale);
  d[lane + 96] = f2bf(o3 * outscale);
}

// ---------- V convert: f32 [tok, kvh*128+d] -> bf16 [b,kvh,s,d] ----------
__global__ void v_convert(const float* __restrict__ Vf, bf16* __restrict__ Vb) {
  int idx = blockIdx.x * blockDim.x + threadIdx.x;  // over MROWS*512
  int d   = idx & 127;
  int kvh = (idx >> 7) & 3;
  int tok = idx >> 9;
  int b = tok / SEQ, s = tok % SEQ;
  Vb[((size_t)(b * NUM_KV_HEADS + kvh) * SEQ + s) * HEAD_DIM + d] = f2bf(Vf[idx]);
}

// ---------- Flash attention with TDM double-buffered K/V staging ----------
// grid = (B*NUM_HEADS, SEQ/128); block = 256 (8 waves), each wave a 16-row q-tile.
__global__ __launch_bounds__(256) void flash_attn(const bf16* __restrict__ Qb_,
                                                  const bf16* __restrict__ Kb_,
                                                  const bf16* __restrict__ Vb_,
                                                  bf16* __restrict__ Obf) {
  __shared__ __align__(16) bf16 sK[2][32 * 128];   // [buf][key][d]  (TDM dest)
  __shared__ __align__(16) bf16 sV[2][32 * 128];   // [buf][key][d]  (TDM dest)
  __shared__ __align__(16) bf16 sVT[128 * 32];     // [d][key]
  __shared__ __align__(16) float sP[8][16 * 32];   // per-wave P scratch [q][key]

  int bh = blockIdx.x;
  int b = bh >> 4, h = bh & 15;
  int kvh = h >> 2;
  int wv  = threadIdx.x >> 5;
  int lane = threadIdx.x & 31;
  int l  = lane & 15;
  int hf = lane >> 4;
  int q0 = blockIdx.y * 128 + wv * 16;

  const bf16* Qp = Qb_ + (size_t)(b * NUM_HEADS + h) * SEQ * HEAD_DIM;
  const bf16* Kp = Kb_ + (size_t)(b * NUM_KV_HEADS + kvh) * SEQ * HEAD_DIM;
  const bf16* Vp = Vb_ + (size_t)(b * NUM_KV_HEADS + kvh) * SEQ * HEAD_DIM;

  unsigned ldsK0 = (unsigned)(size_t)&sK[0][0];
  unsigned ldsK1 = (unsigned)(size_t)&sK[1][0];
  unsigned ldsV0 = (unsigned)(size_t)&sV[0][0];
  unsigned ldsV1 = (unsigned)(size_t)&sV[1][0];

  // preload Q A-frags (16x128 split into four 16x32 K-steps)
  v16bf aQ[4];
#pragma unroll
  for (int ks = 0; ks < 4; ++ks) {
    const bf16* p = Qp + (size_t)(q0 + l) * 128 + ks * 32 + hf * 8;
    aQ[ks] = load_frag(p, p + 16);
  }

  v8f acc[8] = {};
  float mrow[8], lrow[8];
#pragma unroll
  for (int i = 0; i < 8; ++i) { mrow[i] = -1e30f; lrow[i] = 0.f; }

  const int NT = SEQ / 32;
  if (wv == 0) {            // prologue: DMA tile 0 into buffer 0
    tdm_load_tile_32x128(Kp, ldsK0);
    tdm_load_tile_32x128(Vp, ldsV0);
  }

  for (int t = 0; t < NT; ++t) {
    int buf = t & 1;
    if (wv == 0) {
      if (t + 1 < NT) {     // prefetch tile t+1 into other buffer, then wait for tile t
        const bf16* kn = Kp + (size_t)(t + 1) * 32 * 128;
        const bf16* vn = Vp + (size_t)(t + 1) * 32 * 128;
        tdm_load_tile_32x128(kn, buf ? ldsK0 : ldsK1);
        tdm_load_tile_32x128(vn, buf ? ldsV0 : ldsV1);
        __builtin_amdgcn_s_wait_tensorcnt(2);
      } else {
        __builtin_amdgcn_s_wait_tensorcnt(0);
      }
    }
    __syncthreads();        // tile t visible to all waves

    {   // LDS->LDS transpose of V tile: sV[buf][key][d] -> sVT[d][key]
      int key = threadIdx.x >> 3;
      int c   = (threadIdx.x & 7) * 16;
      const bf16* vs = &sV[buf][key * 128 + c];
      v8bf x0 = *(const v8bf*)vs;
      v8bf x1 = *(const v8bf*)(vs + 8);
#pragma unroll
      for (int e = 0; e < 8; ++e) {
        sVT[(c + e) * 32 + key]     = x0[e];
        sVT[(c + 8 + e) * 32 + key] = x1[e];
      }
    }
    __syncthreads();

    // scores: two 16x16 frags over 32 keys
    v8f s0 = {}, s1 = {};
    const bf16* sKb = &sK[buf][0];
#pragma unroll
    for (int ks = 0; ks < 4; ++ks) {
      int kb = ks * 32 + hf * 8;
      const bf16* pk0 = sKb + (0 + l) * 128 + kb;
      const bf16* pk1 = sKb + (16 + l) * 128 + kb;
      v16bf b0 = load_frag(pk0, pk0 + 16);
      v16bf b1 = load_frag(pk1, pk1 + 16);
      s0 = __builtin_amdgcn_wmma_f32_16x16x32_bf16(false, aQ[ks], false, b0, (short)0, s0, false, false);
      s1 = __builtin_amdgcn_wmma_f32_16x16x32_bf16(false, aQ[ks], false, b1, (short)0, s1, false, false);
    }

    // online softmax (row-wise over 16 lanes of each half)
    float p0[8], p1[8];
#pragma unroll
    for (int i = 0; i < 8; ++i) {
      float mx = fmaxf(s0[i], s1[i]);
      mx = fmaxf(mx, __shfl_xor(mx, 1, 32));
      mx = fmaxf(mx, __shfl_xor(mx, 2, 32));
      mx = fmaxf(mx, __shfl_xor(mx, 4, 32));
      mx = fmaxf(mx, __shfl_xor(mx, 8, 32));
      float mnew  = fmaxf(mrow[i], mx);
      float alpha = __expf(mrow[i] - mnew);
      mrow[i] = mnew;
      float e0 = __expf(s0[i] - mnew);
      float e1 = __expf(s1[i] - mnew);
      p0[i] = e0; p1[i] = e1;
      float r = e0 + e1;
      r += __shfl_xor(r, 1, 32);
      r += __shfl_xor(r, 2, 32);
      r += __shfl_xor(r, 4, 32);
      r += __shfl_xor(r, 8, 32);
      lrow[i] = lrow[i] * alpha + r;
#pragma unroll
      for (int f = 0; f < 8; ++f) acc[f][i] *= alpha;
    }

    // C-layout -> A-layout for P via per-wave LDS round trip
    float* Pw = sP[wv];
#pragma unroll
    for (int i = 0; i < 8; ++i) {
      int row = i + hf * 8;
      Pw[row * 32 + l]      = p0[i];
      Pw[row * 32 + 16 + l] = p1[i];
    }
    asm volatile("s_wait_dscnt 0x0" ::: "memory");  // intra-wave LDS RAW fence
    __builtin_amdgcn_wave_barrier();

    v16bf aP;
    {
      const float* pr = Pw + l * 32 + hf * 8;
#pragma unroll
      for (int e = 0; e < 8; ++e) { aP[e] = f2bf(pr[e]); aP[e + 8] = f2bf(pr[16 + e]); }
    }

    // PV: acc[f] += P(16x32) * V(32x16) over 8 d-frags
#pragma unroll
    for (int f = 0; f < 8; ++f) {
      const bf16* pv = sVT + (f * 16 + l) * 32 + hf * 8;
      v16bf bV = load_frag(pv, pv + 16);
      acc[f] = __builtin_amdgcn_wmma_f32_16x16x32_bf16(false, aP, false, bV, (short)0, acc[f], false, false);
    }
    __syncthreads();        // tile-t reads done before next TDM overwrite
  }

  // epilogue: normalize and write [tok, h*128+d] bf16
#pragma unroll
  for (int i = 0; i < 8; ++i) {
    float inv = 1.0f / lrow[i];
    int qrow = q0 + i + hf * 8;
    bf16* od = Obf + ((size_t)(b * SEQ + qrow)) * (NUM_HEADS * HEAD_DIM) + h * HEAD_DIM;
#pragma unroll
    for (int f = 0; f < 8; ++f) od[f * 16 + l] = f2bf(acc[f][i] * inv);
  }
}

// ---------- host ----------
extern "C" void kernel_launch(void* const* d_in, const int* in_sizes, int n_in,
                              void* d_out, int out_size, void* d_ws, size_t ws_size,
                              hipStream_t stream) {
  const float* hidden = (const float*)d_in[0];
  const float* cosp   = (const float*)d_in[1];
  const float* sinp   = (const float*)d_in[2];
  const float* Wq     = (const float*)d_in[3];
  const float* Wk     = (const float*)d_in[4];
  const float* Wv     = (const float*)d_in[5];
  const float* Wo     = (const float*)d_in[6];
  const float* qnw    = (const float*)d_in[7];
  const float* knw    = (const float*)d_in[8];

  size_t off = 0;
  char* ws = (char*)d_ws;
  auto take = [&](size_t bytes) -> void* {
    void* p = ws + off;
    off += (bytes + 255) & ~(size_t)255;
    return p;
  };
  bf16* Xbf  = (bf16*)take((size_t)MROWS * HID * 2);
  bf16* Wqbf = (bf16*)take((size_t)2048 * HID * 2);
  bf16* Wkbf = (bf16*)take((size_t)512 * HID * 2);
  bf16* Wvbf = (bf16*)take((size_t)512 * HID * 2);
  bf16* Wobf = (bf16*)take((size_t)2048 * HID * 2);
  float* Qf  = (float*)take((size_t)MROWS * 2048 * 4);
  float* Kf  = (float*)take((size_t)MROWS * 512 * 4);
  float* Vf  = (float*)take((size_t)MROWS * 512 * 4);
  bf16* Qbf  = (bf16*)take((size_t)MROWS * 2048 * 2);
  bf16* Kbf  = (bf16*)take((size_t)MROWS * 512 * 2);
  bf16* Vbf  = (bf16*)take((size_t)MROWS * 512 * 2);
  bf16* Obf  = (bf16*)take((size_t)MROWS * 2048 * 2);

  // 1. convert inputs to bf16
  int nX = MROWS * HID;
  cvt_f32_bf16<<<(nX + 255) / 256, 256, 0, stream>>>(hidden, Xbf, nX);
  cvt_f32_bf16<<<(2048 * HID + 255) / 256, 256, 0, stream>>>(Wq, Wqbf, 2048 * HID);
  cvt_f32_bf16<<<(512 * HID + 255) / 256, 256, 0, stream>>>(Wk, Wkbf, 512 * HID);
  cvt_f32_bf16<<<(512 * HID + 255) / 256, 256, 0, stream>>>(Wv, Wvbf, 512 * HID);
  cvt_f32_bf16<<<(2048 * HID + 255) / 256, 256, 0, stream>>>(Wo, Wobf, 2048 * HID);

  // 2. projections (waves = (M/64)*(N/64); blocks = waves/8)
  gemm_bf16<<<(MROWS / 64) * (2048 / 64) / 8, 256, 0, stream>>>(Xbf, Wqbf, Qf, MROWS, 2048, HID);
  gemm_bf16<<<(MROWS / 64) * (512 / 64) / 8, 256, 0, stream>>>(Xbf, Wkbf, Kf, MROWS, 512, HID);
  gemm_bf16<<<(MROWS / 64) * (512 / 64) / 8, 256, 0, stream>>>(Xbf, Wvbf, Vf, MROWS, 512, HID);

  // 3. RMSNorm + RoPE; fold 1/sqrt(128) into Q
  const float scale = 0.08838834764831845f;
  norm_rope<<<(MROWS * NUM_HEADS) * 32 / 256, 256, 0, stream>>>(Qf, qnw, cosp, sinp, Qbf, NUM_HEADS, scale);
  norm_rope<<<(MROWS * NUM_KV_HEADS) * 32 / 256, 256, 0, stream>>>(Kf, knw, cosp, sinp, Kbf, NUM_KV_HEADS, 1.0f);
  v_convert<<<(MROWS * 512) / 256, 256, 0, stream>>>(Vf, Vbf);

  // 4. flash attention (TDM-staged K/V, double-buffered)
  dim3 fgrid(BATCH * NUM_HEADS, SEQ / 128);
  flash_attn<<<fgrid, 256, 0, stream>>>(Qbf, Kbf, Vbf, Obf);

  // 5. output projection -> d_out (f32)
  gemm_bf16<<<(MROWS / 64) * (2048 / 64) / 8, 256, 0, stream>>>(Obf, Wobf, (float*)d_out, MROWS, 2048, HID);
}
